// AuM_8641474200082
// MI455X (gfx1250) — compile-verified, matrix operand order checked by reading, not run on hardware
//
#include <hip/hip_runtime.h>
#include <hip/hip_bf16.h>
#include <math.h>

#define B_    2
#define S_    1000
#define DIM   1000
#define DCONV 3
#define NST   16
#define DTR   32

typedef __attribute__((ext_vector_type(16))) __bf16 v16bf;
typedef __attribute__((ext_vector_type(8)))  __bf16 v8bf;
typedef __attribute__((ext_vector_type(8)))  float  v8f;

#if defined(__HIP_DEVICE_COMPILE__) && \
    __has_builtin(__builtin_amdgcn_global_load_async_to_lds_b128) && \
    __has_builtin(__builtin_amdgcn_s_wait_asynccnt)
#define USE_ASYNC_LDS 1
// exact parameter types per hipcc diagnostic: vector_size(16) int, AS1 / AS3
typedef int v4i_vs __attribute__((vector_size(16)));
typedef v4i_vs __attribute__((address_space(1)))* async_gptr;
typedef v4i_vs __attribute__((address_space(3)))* async_lptr;
#else
#define USE_ASYNC_LDS 0
#endif

// ---- bf16 helpers (round-to-nearest-even) ----
__device__ __forceinline__ __bf16 f2bf(float f) {
  unsigned u = __builtin_bit_cast(unsigned, f);
  unsigned r = (u + 0x7FFFu + ((u >> 16) & 1u)) >> 16;
  unsigned short s = (unsigned short)r;
  return __builtin_bit_cast(__bf16, s);
}
__device__ __forceinline__ float bf2f(__bf16 b) {
  unsigned short s = __builtin_bit_cast(unsigned short, b);
  unsigned u = ((unsigned)s) << 16;
  return __builtin_bit_cast(float, u);
}

// ---------------------------------------------------------------------------
// fp32 -> padded bf16 hi/lo planes.  src element (r,c) at src[r*ldr + c*cstep].
// ---------------------------------------------------------------------------
__global__ void split_plane(const float* __restrict__ src, int R, int C,
                            int ldr, int cstep,
                            __bf16* __restrict__ hi, __bf16* __restrict__ lo,
                            int Rp, int Cp) {
  int idx = blockIdx.x * blockDim.x + threadIdx.x;
  if (idx >= Rp * Cp) return;
  int r = idx / Cp, c = idx - r * Cp;
  float v = (r < R && c < C) ? src[(size_t)r * ldr + (size_t)c * cstep] : 0.f;
  __bf16 h = f2bf(v);
  hi[idx] = h;
  lo[idx] = f2bf(v - bf2f(h));
}

// Transposed split for the conv B-matrix: plane row r = shifted column h=r-1.
__global__ void split_plane_T(const float* __restrict__ src, int K, int N,
                              __bf16* __restrict__ hi, __bf16* __restrict__ lo,
                              int Rp, int Cp) {
  int b = blockIdx.y;
  int idx = blockIdx.x * blockDim.x + threadIdx.x;
  if (idx >= Rp * Cp) return;
  int r = idx / Cp, c = idx - r * Cp;
  int h = r - 1;
  const float* s = src + (size_t)b * K * N;
  float v = (h >= 0 && h < N && c < K) ? s[(size_t)c * N + h] : 0.f;
  size_t o = (size_t)b * Rp * Cp + idx;
  __bf16 hh = f2bf(v);
  hi[o] = hh;
  lo[o] = f2bf(v - bf2f(hh));
}

// Fragment gather from a per-lane base pointer + compile-time-friendly offset:
// two contiguous 16B runs (ISA 7.12.2 bf16 A/B layout).
__device__ __forceinline__ v16bf frag_at(const __bf16* p) {
  v8bf lo = *(const v8bf*)(p);
  v8bf hi = *(const v8bf*)(p + 16);
  return __builtin_shufflevector(lo, hi, 0, 1, 2, 3, 4, 5, 6, 7,
                                 8, 9, 10, 11, 12, 13, 14, 15);
}

#define WMMA_BF16(A, B, Cc) \
  __builtin_amdgcn_wmma_f32_16x16x32_bf16(false, (A), false, (B), (short)0, (Cc), false, false)

// ---------------------------------------------------------------------------
// C = act(A * W^T + bias), 32x64 macro-tile per wave (2 M-tiles x 4 N-tiles).
// LK is a compile-time leading dimension so all fragment addresses fold into
// immediate offsets from one base pointer per plane (no per-fragment address
// VGPR pairs, no spills).  ACT: 0=none 1=silu 2=softplus
// ---------------------------------------------------------------------------
template <int ACT, int LK>
__global__ void gemm_bf16(const __bf16* __restrict__ Ah, const __bf16* __restrict__ Al,
                          const __bf16* __restrict__ Wh, const __bf16* __restrict__ Wl,
                          const float* __restrict__ bias,
                          float* __restrict__ C, int ldc,
                          int M, int N, int Kp, int pairsM, int groupsN) {
  int wid = blockIdx.x * (blockDim.x >> 5) + (threadIdx.x >> 5);
  if (wid >= pairsM * groupsN) return;
  int pm = wid / groupsN, gn = wid - pm * groupsN;
  int lane = threadIdx.x & 31;
  int rc = lane & 15;
  int koff = (lane >> 4) << 3;
  int m0 = pm << 5, n0 = gn << 6;

  // one per-lane base pointer per plane; everything else = constant offsets
  const __bf16* pa_h = Ah + (size_t)(m0 + rc) * LK + koff;
  const __bf16* pa_l = Al + (size_t)(m0 + rc) * LK + koff;
  const __bf16* pw_h = Wh + (size_t)(n0 + rc) * LK + koff;
  const __bf16* pw_l = Wl + (size_t)(n0 + rc) * LK + koff;

  v8f acc[2][4] = {{{}, {}, {}, {}}, {{}, {}, {}, {}}};
  for (int k0 = 0; k0 < Kp; k0 += 32) {
    __builtin_prefetch(pa_h + k0 + 64, 0, 3);
    v16bf a0h = frag_at(pa_h + k0);
    v16bf a0l = frag_at(pa_l + k0);
    v16bf a1h = frag_at(pa_h + 16 * LK + k0);
    v16bf a1l = frag_at(pa_l + 16 * LK + k0);
#pragma unroll
    for (int t = 0; t < 4; ++t) {
      v16bf b_h = frag_at(pw_h + t * 16 * LK + k0);
      v16bf b_l = frag_at(pw_l + t * 16 * LK + k0);
      acc[0][t] = WMMA_BF16(a0h, b_h, acc[0][t]);
      acc[1][t] = WMMA_BF16(a1h, b_h, acc[1][t]);
      acc[0][t] = WMMA_BF16(a0h, b_l, acc[0][t]);
      acc[1][t] = WMMA_BF16(a1h, b_l, acc[1][t]);
      acc[0][t] = WMMA_BF16(a0l, b_h, acc[0][t]);
      acc[1][t] = WMMA_BF16(a1l, b_h, acc[1][t]);
    }
  }

#pragma unroll
  for (int u = 0; u < 2; ++u) {
    int mb = m0 + (u << 4) + ((lane >> 4) << 3);
#pragma unroll
    for (int t = 0; t < 4; ++t) {
      int n = n0 + (t << 4) + (lane & 15);
      if (n < N) {
        float bv = bias ? bias[n] : 0.f;
#pragma unroll
        for (int r = 0; r < 8; ++r) {
          int m = mb + r;
          if (m < M) {
            float v = acc[u][t][r] + bv;
            if (ACT == 1) v = v / (1.f + __expf(-v));                   // silu
            else if (ACT == 2) v = (v > 20.f) ? v : log1pf(__expf(v));  // softplus
            C[(size_t)m * ldc + n] = v;
          }
        }
      }
    }
  }
}

// ---------------------------------------------------------------------------
// Conv = 3 shifted GEMMs folded into the K loop (taps = row offsets into the
// pre-transposed +1-shifted B planes); 32x64 macro-tile; silu epilogue.
// ---------------------------------------------------------------------------
__global__ void conv_gemm_bf16(const __bf16* __restrict__ CWh, const __bf16* __restrict__ CWl,
                               const __bf16* __restrict__ BTh, const __bf16* __restrict__ BTl,
                               float* __restrict__ P2) {
  const int pairsM = 32, groupsN = 16;       // M=N=1000
  const int LK = 1024;
  const int TAP = 1024 * 1024, BS = 1056 * 1024;
  int b = blockIdx.z;
  int wid = blockIdx.x * (blockDim.x >> 5) + (threadIdx.x >> 5);
  if (wid >= pairsM * groupsN) return;
  int pm = wid / groupsN, gn = wid - pm * groupsN;
  int lane = threadIdx.x & 31;
  int rc = lane & 15;
  int koff = (lane >> 4) << 3;
  int m0 = pm << 5, n0 = gn << 6;

  const __bf16* pa_h = CWh + (size_t)(m0 + rc) * LK + koff;   // +tap*TAP const
  const __bf16* pa_l = CWl + (size_t)(m0 + rc) * LK + koff;
  const __bf16* pb_h = BTh + (size_t)b * BS + (size_t)(n0 + rc) * LK + koff; // +tap*LK const
  const __bf16* pb_l = BTl + (size_t)b * BS + (size_t)(n0 + rc) * LK + koff;

  v8f acc[2][4] = {{{}, {}, {}, {}}, {{}, {}, {}, {}}};
  for (int k0 = 0; k0 < 1024; k0 += 32) {
#pragma unroll
    for (int tap = 0; tap < DCONV; ++tap) {
      v16bf a0h = frag_at(pa_h + tap * TAP + k0);
      v16bf a0l = frag_at(pa_l + tap * TAP + k0);
      v16bf a1h = frag_at(pa_h + tap * TAP + 16 * LK + k0);
      v16bf a1l = frag_at(pa_l + tap * TAP + 16 * LK + k0);
#pragma unroll
      for (int t = 0; t < 4; ++t) {
        v16bf b_h = frag_at(pb_h + (t * 16 + tap) * LK + k0);
        v16bf b_l = frag_at(pb_l + (t * 16 + tap) * LK + k0);
        acc[0][t] = WMMA_BF16(a0h, b_h, acc[0][t]);
        acc[1][t] = WMMA_BF16(a1h, b_h, acc[1][t]);
        acc[0][t] = WMMA_BF16(a0h, b_l, acc[0][t]);
        acc[1][t] = WMMA_BF16(a1h, b_l, acc[1][t]);
        acc[0][t] = WMMA_BF16(a0l, b_h, acc[0][t]);
        acc[1][t] = WMMA_BF16(a1l, b_h, acc[1][t]);
      }
    }
  }

  float* out = P2 + (size_t)b * S_ * DIM;
#pragma unroll
  for (int u = 0; u < 2; ++u) {
    int mb = m0 + (u << 4) + ((lane >> 4) << 3);
#pragma unroll
    for (int t = 0; t < 4; ++t) {
      int n = n0 + (t << 4) + (lane & 15);
      if (n < DIM) {
#pragma unroll
        for (int r = 0; r < 8; ++r) {
          int m = mb + r;
          if (m < DIM) {
            float v = acc[u][t][r];
            out[(size_t)m * DIM + n] = v / (1.f + __expf(-v));   // silu
          }
        }
      }
    }
  }
}

// ---------------------------------------------------------------------------
// Sequential selective scan; per-step B/C vectors staged into LDS via the
// CDNA5 async global->LDS path (b128 per lane) + s_wait_asynccnt.
// ---------------------------------------------------------------------------
__global__ void scan_kernel(const float* __restrict__ X2,    // path2 (B,S,DIM)
                            const float* __restrict__ Del,   // delta (B,S,DIM)
                            const float* __restrict__ dBC,   // (B,S,64)
                            const float* __restrict__ A_log, // (DIM,NST)
                            const float* __restrict__ Dp,    // (DIM)
                            float* __restrict__ Y) {         // (B,S,DIM)
  __shared__ float sBC[2 * NST];
  int b = blockIdx.y;
  int d = blockIdx.x * blockDim.x + threadIdx.x;
  bool act = d < DIM;

  float Aa[NST], h[NST];
  float Dpd = 0.f;
  if (act) {
    Dpd = Dp[d];
#pragma unroll
    for (int n = 0; n < NST; ++n) {
      Aa[n] = -__expf(A_log[(size_t)d * NST + n]);
      h[n] = 0.f;
    }
  }
  const size_t bs = (size_t)b * S_ * DIM;
  const size_t bc = (size_t)b * S_ * 64;

  for (int s = 0; s < S_; ++s) {
#if USE_ASYNC_LDS
    if (threadIdx.x < 8) {
      const float* g = dBC + bc + (size_t)s * 64 + DTR + threadIdx.x * 4;
      __builtin_amdgcn_global_load_async_to_lds_b128(
          (async_gptr)(uintptr_t)g,
          (async_lptr)(unsigned)(uintptr_t)(sBC + threadIdx.x * 4),
          0, 0);
    }
    __builtin_amdgcn_s_wait_asynccnt(0);
    __syncthreads();
#else
    if (threadIdx.x < 2 * NST)
      sBC[threadIdx.x] = dBC[bc + (size_t)s * 64 + DTR + threadIdx.x];
    __syncthreads();
#endif
    if (act) {
      float dlt = Del[bs + (size_t)s * DIM + d];
      float x2  = X2[bs + (size_t)s * DIM + d];
      float y = 0.f;
#pragma unroll
      for (int n = 0; n < NST; ++n) {
        float dA = __expf(dlt * Aa[n]);
        h[n] = dA * h[n] + (dlt * sBC[n]) * x2;
        y += h[n] * sBC[NST + n];
      }
      Y[bs + (size_t)s * DIM + d] = y + Dpd * x2;
    }
    __syncthreads();
  }
}

// y = silu(p1) * y   (in place)
__global__ void gate_mult(const float* __restrict__ P1, float* __restrict__ Y, int n) {
  int i = blockIdx.x * blockDim.x + threadIdx.x;
  if (i < n) {
    float a = P1[i];
    Y[i] = (a / (1.f + __expf(-a))) * Y[i];
  }
}

// ---------------------------------------------------------------------------
extern "C" void kernel_launch(void* const* d_in, const int* in_sizes, int n_in,
                              void* d_out, int out_size, void* d_ws, size_t ws_size,
                              hipStream_t stream) {
  const float* x         = (const float*)d_in[0];
  const float* proj_w    = (const float*)d_in[1];
  const float* proj_b    = (const float*)d_in[2];
  const float* conv_w    = (const float*)d_in[3];
  const float* deltaBC_w = (const float*)d_in[4];
  const float* dt_proj_w = (const float*)d_in[5];
  const float* dt_proj_b = (const float*)d_in[6];
  const float* A_log     = (const float*)d_in[7];
  const float* Dp        = (const float*)d_in[8];
  float* out = (float*)d_out;

  // ---- workspace layout (256B aligned chunks) ----
  size_t off = 0;
  char* base = (char*)d_ws;
  auto alloc = [&](size_t bytes) -> char* {
    char* p = base + off;
    off = (off + bytes + 255) & ~(size_t)255;
    return p;
  };
  const size_t NE = (size_t)B_ * S_ * DIM;          // 2,000,000
  const int MP = 2016;                              // M=2000 padded to 32-row pairs
  float*  p1f   = (float*)alloc(NE * 4);
  float*  p2f   = (float*)alloc(NE * 4);
  float*  dltf  = (float*)alloc(NE * 4);
  float*  yf    = (float*)alloc(NE * 4);
  float*  dbcf  = (float*)alloc((size_t)B_ * S_ * 64 * 4);
  __bf16* pA_h  = (__bf16*)alloc((size_t)MP * 1024 * 2);   // x / p2 / mult planes (reused)
  __bf16* pA_l  = (__bf16*)alloc((size_t)MP * 1024 * 2);
  __bf16* pw_h  = (__bf16*)alloc((size_t)1024 * 1024 * 2);
  __bf16* pw_l  = (__bf16*)alloc((size_t)1024 * 1024 * 2);
  __bf16* cw_h  = (__bf16*)alloc((size_t)3 * 1024 * 1024 * 2);
  __bf16* cw_l  = (__bf16*)alloc((size_t)3 * 1024 * 1024 * 2);
  __bf16* bt_h  = (__bf16*)alloc((size_t)B_ * 1056 * 1024 * 2); // p1^T, +1 shift
  __bf16* bt_l  = (__bf16*)alloc((size_t)B_ * 1056 * 1024 * 2);
  __bf16* dbw_h = (__bf16*)alloc((size_t)64 * 1024 * 2);
  __bf16* dbw_l = (__bf16*)alloc((size_t)64 * 1024 * 2);
  __bf16* dtp_h = (__bf16*)alloc((size_t)1024 * 32 * 2);
  __bf16* dtp_l = (__bf16*)alloc((size_t)1024 * 32 * 2);
  __bf16* dbc_h = (__bf16*)alloc((size_t)MP * 32 * 2);
  __bf16* dbc_l = (__bf16*)alloc((size_t)MP * 32 * 2);

  const int M = B_ * S_;              // 2000
  const int pM = MP / 32;             // 63 M-pairs
  auto blk = [](int waves) { return (waves + 3) / 4; };

  // ---- operand splits (O(N^2), L2-resident) ----
  split_plane<<<(MP * 1024 + 255) / 256, 256, 0, stream>>>(x, M, DIM, DIM, 1, pA_h, pA_l, MP, 1024);
  split_plane<<<(1024 * 1024 + 255) / 256, 256, 0, stream>>>(proj_w, DIM, DIM, DIM, 1, pw_h, pw_l, 1024, 1024);
  for (int t = 0; t < DCONV; ++t)
    split_plane<<<(1024 * 1024 + 255) / 256, 256, 0, stream>>>(
        conv_w + t, DIM, S_, S_ * DCONV, DCONV,
        cw_h + (size_t)t * 1024 * 1024, cw_l + (size_t)t * 1024 * 1024, 1024, 1024);
  split_plane<<<(64 * 1024 + 255) / 256, 256, 0, stream>>>(deltaBC_w, DTR + 2 * NST, DIM, DIM, 1, dbw_h, dbw_l, 64, 1024);
  split_plane<<<(1024 * 32 + 255) / 256, 256, 0, stream>>>(dt_proj_w, DIM, DTR, DTR, 1, dtp_h, dtp_l, 1024, 32);

  // 1) p1 = x @ proj_w^T + proj_b
  gemm_bf16<0, 1024><<<blk(pM * 16), 128, 0, stream>>>(pA_h, pA_l, pw_h, pw_l,
                                                       proj_b, p1f, DIM, M, DIM, 1024, pM, 16);
  // 2) transpose+shift+split p1 per batch, then conv+silu -> p2
  {
    dim3 g((1056 * 1024 + 255) / 256, B_);
    split_plane_T<<<g, 256, 0, stream>>>(p1f, S_, DIM, bt_h, bt_l, 1056, 1024);
  }
  {
    dim3 g(blk(32 * 16), 1, B_);
    conv_gemm_bf16<<<g, 128, 0, stream>>>(cw_h, cw_l, bt_h, bt_l, p2f);
  }
  // 3) dBC = p2 @ deltaBC_w^T   (N=64 -> one 4-subtile group)
  split_plane<<<(MP * 1024 + 255) / 256, 256, 0, stream>>>(p2f, M, DIM, DIM, 1, pA_h, pA_l, MP, 1024);
  gemm_bf16<0, 1024><<<blk(pM * 1), 128, 0, stream>>>(pA_h, pA_l, dbw_h, dbw_l,
                                                      nullptr, dbcf, 64, M, 64, 1024, pM, 1);
  // 4) delta = softplus(dBC[:, :32] @ dt_proj_w^T + dt_proj_b)
  split_plane<<<(MP * 32 + 255) / 256, 256, 0, stream>>>(dbcf, M, DTR, DTR + 2 * NST, 1, dbc_h, dbc_l, MP, 32);
  gemm_bf16<2, 32><<<blk(pM * 16), 128, 0, stream>>>(dbc_h, dbc_l, dtp_h, dtp_l,
                                                     dt_proj_b, dltf, DIM, M, DIM, 32, pM, 16);
  // 5) selective scan -> y
  {
    dim3 g((DIM + 255) / 256, B_);
    scan_kernel<<<g, 256, 0, stream>>>(p2f, dltf, dbcf, A_log, Dp, yf);
  }
  // 6) y = silu(p1) * y
  gate_mult<<<((int)NE + 255) / 256, 256, 0, stream>>>(p1f, yf, (int)NE);
  // 7) out = y @ proj_w^T + proj_b
  split_plane<<<(MP * 1024 + 255) / 256, 256, 0, stream>>>(yf, M, DIM, DIM, 1, pA_h, pA_l, MP, 1024);
  gemm_bf16<0, 1024><<<blk(pM * 16), 128, 0, stream>>>(pA_h, pA_l, pw_h, pw_l,
                                                       proj_b, out, DIM, M, DIM, 1024, pM, 16);
}
